// MoE_22436909154693
// MI455X (gfx1250) — matrix operand order repaired and checked by reading, not compiled, not run on hardware
//
#include <hip/hip_runtime.h>
#include <cstdint>
#include <cstddef>

// ---------------------------------------------------------------------------
// MoE (8 experts, top-2, D=1024, H=128, T=8192) for MI455X / gfx1250.
// bf16 WMMA (v_wmma_f32_16x16x32_bf16) for both expert GEMMs, restructured as
// two regular GEMMs over a concatenated expert axis; gate routing folded in
// as row scaling (zero for non-top-2 experts). LDS tiles are filled with the
// CDNA5 async memory->LDS DMA path (ASYNCcnt) when the toolchain exposes it.
// ---------------------------------------------------------------------------

typedef __bf16 bf16_t;
typedef __attribute__((ext_vector_type(16))) __bf16 v16bf;
typedef __attribute__((ext_vector_type(8)))  __bf16 v8bf;
typedef __attribute__((ext_vector_type(4)))  __bf16 v4bf;
typedef __attribute__((ext_vector_type(8)))  float  v8f;

union FragAB { v16bf v; v8bf h[2]; };

#define NTOK   8192
#define DDIM   1024
#define HDIM   128
#define NEXP   8

// LDS tile row padded 32 -> 48 bf16 (96 B = 24 banks; gcd(24,64)=8 => 2-way
// max conflict on 16-lane ds_load_b128 fragment reads; 16B alignment kept).
#define KPAD   48

// ---- CDNA5 async memory->LDS staging (guarded; falls back to reg staging) --
#if defined(__AMDGCN__) && __has_builtin(__builtin_amdgcn_global_load_async_to_lds_b128)
  #define USE_ASYNC_LDS 1
#else
  #define USE_ASYNC_LDS 0
#endif

#if USE_ASYNC_LDS
  // Builtin signature (from hipcc diagnostics): param 0 is
  // 'int __vector(4) addrspace(1)*' (printed as '__device__'), param 1 is the
  // LDS-side pointer. Cast to typed int4 pointers in the right address spaces.
  typedef int v4i_t __attribute__((vector_size(4 * sizeof(int))));
  typedef __attribute__((address_space(1))) v4i_t* as1_v4i;
  typedef __attribute__((address_space(3))) v4i_t* as3_v4i;
  // global pointer: flat address bits == global address bits -> via uintptr_t.
  #define GPTR(p) ((as1_v4i)(uintptr_t)(p))
  // LDS pointer: proper addrspacecast (strips flat aperture bits).
  #define LPTR(p) ((as3_v4i)(p))
  #if __has_builtin(__builtin_amdgcn_s_wait_asynccnt)
    #define WAIT_ASYNC0() __builtin_amdgcn_s_wait_asynccnt(0)
  #else
    #define WAIT_ASYNC0() asm volatile("s_wait_asynccnt 0x0" ::: "memory")
  #endif
#endif

// ---------------------------------------------------------------------------
// f32 -> bf16 conversion kernels (one-shot staging; bandwidth-trivial).
// ---------------------------------------------------------------------------
__global__ __launch_bounds__(256)
void cvt_x_kernel(const float* __restrict__ x, bf16_t* __restrict__ Xb) {
    const size_t i = ((size_t)blockIdx.x * 256 + threadIdx.x) * 4;
    const float4 vv = *(const float4*)(x + i);
    v4bf o;
    o[0] = (bf16_t)vv.x; o[1] = (bf16_t)vv.y;
    o[2] = (bf16_t)vv.z; o[3] = (bf16_t)vv.w;
    *(v4bf*)(Xb + i) = o;
}

// W1t[e][h][k] = bf16(w1[e][k][h])   (transpose so LDS staging is k-contiguous)
__global__ __launch_bounds__(256)
void cvt_w1_kernel(const float* __restrict__ w1, bf16_t* __restrict__ W1t) {
    const int i = blockIdx.x * 256 + threadIdx.x;     // [e][h][k]
    const int e = i >> 17;
    const int r = i & 131071;
    const int h = r >> 10;
    const int k = r & 1023;
    W1t[i] = (bf16_t)w1[(size_t)e * 131072 + (size_t)k * 128 + h];
}

// W2t[o][e*128+h] = bf16(w2[e][h][o])  (concat-K transposed layout)
__global__ __launch_bounds__(256)
void cvt_w2_kernel(const float* __restrict__ w2, bf16_t* __restrict__ W2t) {
    const int i = blockIdx.x * 256 + threadIdx.x;     // [o][e*128+h]
    const int o  = i >> 10;
    const int kk = i & 1023;
    const int e  = kk >> 7;
    const int h  = kk & 127;
    W2t[i] = (bf16_t)w2[(size_t)e * 131072 + (size_t)h * 1024 + o];
}

// ---------------------------------------------------------------------------
// Gate: one wave32 per token. logits = x@gate_w + gate_b, softmax, top-2.
// Writes dense Wg[t][e] = softmax weight if e in top-2 else 0.
// ---------------------------------------------------------------------------
__global__ __launch_bounds__(256)
void gate_kernel(const float* __restrict__ x, const float* __restrict__ gw,
                 const float* __restrict__ gb, float* __restrict__ Wg)
{
    const int lane = threadIdx.x & 31;
    const int wid  = threadIdx.x >> 5;
    const int t    = blockIdx.x * 8 + wid;
    const float* xr = x + (size_t)t * DDIM;

    float acc[8] = {0.f, 0.f, 0.f, 0.f, 0.f, 0.f, 0.f, 0.f};
    for (int it = 0; it < DDIM / 32; ++it) {
        const int d = it * 32 + lane;
        const float xv  = xr[d];
        const float4 g0 = *(const float4*)(gw + d * 8);
        const float4 g1 = *(const float4*)(gw + d * 8 + 4);
        acc[0] = fmaf(xv, g0.x, acc[0]); acc[1] = fmaf(xv, g0.y, acc[1]);
        acc[2] = fmaf(xv, g0.z, acc[2]); acc[3] = fmaf(xv, g0.w, acc[3]);
        acc[4] = fmaf(xv, g1.x, acc[4]); acc[5] = fmaf(xv, g1.y, acc[5]);
        acc[6] = fmaf(xv, g1.z, acc[6]); acc[7] = fmaf(xv, g1.w, acc[7]);
    }
    // wave32 butterfly reduction
    #pragma unroll
    for (int e = 0; e < 8; ++e) {
        #pragma unroll
        for (int off = 16; off > 0; off >>= 1)
            acc[e] += __shfl_xor(acc[e], off, 32);
    }
    float lgt[8], mx = -3.4e38f;
    #pragma unroll
    for (int e = 0; e < 8; ++e) { lgt[e] = acc[e] + gb[e]; mx = fmaxf(mx, lgt[e]); }
    float p[8], s = 0.f;
    #pragma unroll
    for (int e = 0; e < 8; ++e) { p[e] = __expf(lgt[e] - mx); s += p[e]; }
    const float inv = 1.0f / s;
    int i0 = 0; float b0 = p[0];
    #pragma unroll
    for (int e = 1; e < 8; ++e) if (p[e] > b0) { b0 = p[e]; i0 = e; }
    int i1 = -1; float b1v = -1.f;
    #pragma unroll
    for (int e = 0; e < 8; ++e) if (e != i0 && p[e] > b1v) { b1v = p[e]; i1 = e; }
    if (lane == 0) {
        #pragma unroll
        for (int e = 0; e < 8; ++e)
            Wg[t * 8 + e] = (e == i0 || e == i1) ? p[e] * inv : 0.0f;
    }
}

// ---------------------------------------------------------------------------
// WMMA GEMM: C[128 x 128] tile per workgroup of 512 threads (16 waves).
// Each wave owns a 16x64 strip: acc = 4 tiles (32 VGPRs), A frag shared over
// 4 WMMAs; per-wave VGPR pressure ~90 so accumulators stay pinned in place.
// K = 1024, K-step 32, double-buffered padded LDS. Tile ks+1 is DMA'd into
// the other buffer (global_load_async_to_lds_b128, ASYNCcnt) while tile ks is
// consumed by the WMMA chain; global_prefetch warms GL2 two steps ahead.
// MODE 1: Hc[t][e*128+n] = Wg[t][e] * relu(acc + b1[e][n])     (bf16 out)
// MODE 2: Out[t][n]      = acc + sum_e Wg[t][e] * b2[e][n]     (f32 out)
// ---------------------------------------------------------------------------
template <int MODE>
__global__ __launch_bounds__(512)
void moe_gemm_wmma(const bf16_t* __restrict__ A,    // [8192 x 1024] row-major
                   const bf16_t* __restrict__ Bt,   // [Ntot x 1024] row-major (W^T)
                   const float*  __restrict__ bias, // b1 [8*128] or b2 [8*1024]
                   const float*  __restrict__ Wg,   // [8192 x 8]
                   bf16_t* __restrict__ Hc,
                   float*  __restrict__ Out)
{
    constexpr int K = 1024;
    constexpr int KSTEPS = K / 32;

    __shared__ bf16_t As[2][128][KPAD];
    __shared__ bf16_t Bs[2][128][KPAD];

    const int tid = threadIdx.x;
    const int m0  = blockIdx.x * 128;
    const int nb  = blockIdx.y;                    // expert (MODE1) / N-tile (MODE2)
    const bf16_t* Asrc = A  + (size_t)m0 * K;
    const bf16_t* Bsrc = Bt + (size_t)nb * 128 * K;

    // Staging: each of 512 threads moves one 16B chunk for A and one for B
    // per K-step (128 rows x 32 k = 4096 bf16 per tile).
    const int lin = tid * 8;
    const int ar  = lin >> 5;
    const int ak  = lin & 31;

#if USE_ASYNC_LDS
    __builtin_amdgcn_global_load_async_to_lds_b128(
        GPTR(Asrc + (size_t)ar * K + ak), LPTR(&As[0][ar][ak]), 0, 0);
    __builtin_amdgcn_global_load_async_to_lds_b128(
        GPTR(Bsrc + (size_t)ar * K + ak), LPTR(&Bs[0][ar][ak]), 0, 0);
    WAIT_ASYNC0();
    __syncthreads();
#else
    uint4 ra, rb;
    ra = *(const uint4*)(Asrc + (size_t)ar * K + ak);
    rb = *(const uint4*)(Bsrc + (size_t)ar * K + ak);
    *(uint4*)&As[0][ar][ak] = ra;
    *(uint4*)&Bs[0][ar][ak] = rb;
    __syncthreads();
#endif

    const int lane  = tid & 31;
    const int wid   = tid >> 5;          // 0..15
    const int rbase = (wid & 7) * 16;    // 16-row strip
    const int cbase = (wid >> 3) * 64;   // 64-col strip
    const int l15   = lane & 15;
    const int lh    = lane >> 4;

    v8f acc[4] = {};

    for (int ks = 0; ks < KSTEPS; ++ks) {
        const int cur  = ks & 1;
        const int nxt  = cur ^ 1;
        const bool more = (ks + 1) < KSTEPS;
        if (more) {
            const int k0 = (ks + 1) * 32;
#if USE_ASYNC_LDS
            // DMA tile ks+1 straight into the idle buffer; overlaps the WMMA
            // chain below (nxt was last read in iter ks-1, barrier-protected).
            __builtin_amdgcn_global_load_async_to_lds_b128(
                GPTR(Asrc + (size_t)ar * K + k0 + ak), LPTR(&As[nxt][ar][ak]), 0, 0);
            __builtin_amdgcn_global_load_async_to_lds_b128(
                GPTR(Bsrc + (size_t)ar * K + k0 + ak), LPTR(&Bs[nxt][ar][ak]), 0, 0);
#else
            ra = *(const uint4*)(Asrc + (size_t)ar * K + k0 + ak);
            rb = *(const uint4*)(Bsrc + (size_t)ar * K + k0 + ak);
#endif
        }
        if (ks + 2 < KSTEPS) {
            const int k2 = (ks + 2) * 32;
            __builtin_prefetch(Asrc + (size_t)ar * K + k2 + ak, 0, 1);
            __builtin_prefetch(Bsrc + (size_t)ar * K + k2 + ak, 0, 1);
        }
        // A fragment: row = rbase + (lane&15); K = kb..kb+7 and kb+16..kb+23,
        // kb = (lane>>4)*8  (ISA 16-bit A layout).
        FragAB a;
        a.h[0] = *(const v8bf*)&As[cur][rbase + l15][lh * 8];
        a.h[1] = *(const v8bf*)&As[cur][rbase + l15][lh * 8 + 16];
        #pragma unroll
        for (int j = 0; j < 4; ++j) {
            // B fragment: col = cbase + j*16 + (lane&15); K = kb..kb+7 and kb+16..kb+23, kb = (lane>>4)*8.
            FragAB b;
            b.h[0] = *(const v8bf*)&Bs[cur][cbase + j * 16 + l15][lh * 8];
            b.h[1] = *(const v8bf*)&Bs[cur][cbase + j * 16 + l15][lh * 8 + 16];
            acc[j] = __builtin_amdgcn_wmma_f32_16x16x32_bf16(
                false, a.v, false, b.v, (short)0, acc[j], false, false);
        }
#if USE_ASYNC_LDS
        if (more) WAIT_ASYNC0();
        __syncthreads();
#else
        if (more) {
            *(uint4*)&As[nxt][ar][ak] = ra;
            *(uint4*)&Bs[nxt][ar][ak] = rb;
        }
        __syncthreads();
#endif
    }

    // C/D layout: lanes 0-15 -> M = v, lanes 16-31 -> M = v + 8; N = lane&15.
    if constexpr (MODE == 1) {
        const float* b1e = bias + nb * HDIM;
        #pragma unroll
        for (int v = 0; v < 8; ++v) {
            const int t = m0 + rbase + lh * 8 + v;
            const float gwv = Wg[t * 8 + nb];
            #pragma unroll
            for (int j = 0; j < 4; ++j) {
                const int n = cbase + j * 16 + l15;
                float val = acc[j][v] + b1e[n];
                val = fmaxf(val, 0.0f) * gwv;
                Hc[(size_t)t * 1024 + nb * HDIM + n] = (bf16_t)val;
            }
        }
    } else {
        const int n0 = nb * 128;
        #pragma unroll
        for (int v = 0; v < 8; ++v) {
            const int t = m0 + rbase + lh * 8 + v;
            const float4 g0 = *(const float4*)(Wg + t * 8);
            const float4 g1 = *(const float4*)(Wg + t * 8 + 4);
            #pragma unroll
            for (int j = 0; j < 4; ++j) {
                const int n = n0 + cbase + j * 16 + l15;
                float bsum = g0.x * bias[n]
                           + g0.y * bias[1024 + n]
                           + g0.z * bias[2048 + n]
                           + g0.w * bias[3072 + n]
                           + g1.x * bias[4096 + n]
                           + g1.y * bias[5120 + n]
                           + g1.z * bias[6144 + n]
                           + g1.w * bias[7168 + n];
                Out[(size_t)t * 1024 + n] = acc[j][v] + bsum;
            }
        }
    }
}

// ---------------------------------------------------------------------------
// Host launcher. Workspace layout (d_ws):
//   [ 0,16M)  Xb   : bf16 x                [8192 x 1024]
//   [16,18M)  W1t  : bf16 w1^T             [8][128][1024]
//   [18,20M)  W2t  : bf16 w2 concat-K^T    [1024][1024]
//   [20,36M)  Hc   : bf16 weighted relu(h) [8192 x 1024]
//   [36M,+256K) Wg : f32 gate weights      [8192 x 8]
// ---------------------------------------------------------------------------
extern "C" void kernel_launch(void* const* d_in, const int* in_sizes, int n_in,
                              void* d_out, int out_size, void* d_ws, size_t ws_size,
                              hipStream_t stream) {
    const float* x      = (const float*)d_in[0];
    const float* gate_w = (const float*)d_in[1];
    const float* gate_b = (const float*)d_in[2];
    const float* w1     = (const float*)d_in[3];
    const float* b1     = (const float*)d_in[4];
    const float* w2     = (const float*)d_in[5];
    const float* b2     = (const float*)d_in[6];
    float* out = (float*)d_out;

    char* ws = (char*)d_ws;
    bf16_t* Xb  = (bf16_t*)(ws);
    bf16_t* W1t = (bf16_t*)(ws + (size_t)(16u << 20));
    bf16_t* W2t = (bf16_t*)(ws + (size_t)(18u << 20));
    bf16_t* Hc  = (bf16_t*)(ws + (size_t)(20u << 20));
    float*  Wg  = (float*) (ws + (size_t)(36u << 20));

    // Stage bf16 operands (x: 8M elems / 4 per thread; weights: 1M elems each).
    cvt_x_kernel <<<8192, 256, 0, stream>>>(x, Xb);
    cvt_w1_kernel<<<4096, 256, 0, stream>>>(w1, W1t);
    cvt_w2_kernel<<<4096, 256, 0, stream>>>(w2, W2t);

    // Gate: 8 tokens per 256-thread block.
    gate_kernel<<<NTOK / 8, 256, 0, stream>>>(x, gate_w, gate_b, Wg);

    // GEMM1: per expert, H = Wg .* relu(x @ w1[e] + b1[e])  -> Hc (bf16)
    moe_gemm_wmma<1><<<dim3(NTOK / 128, NEXP), 512, 0, stream>>>(
        Xb, W1t, b1, Wg, Hc, nullptr);

    // GEMM2: out = Hc @ W2cat + sum_e Wg[:,e]*b2[e]         -> out (f32)
    moe_gemm_wmma<2><<<dim3(NTOK / 128, DDIM / 128), 512, 0, stream>>>(
        Hc, W2t, b2, Wg, nullptr, out);
}
